// BMMRouter_46067819217191
// MI455X (gfx1250) — matrix operand order repaired and checked by reading, not compiled, not run on hardware
//
#include <hip/hip_runtime.h>
#include <hip/hip_bf16.h>

// ---------------- problem constants ----------------
#define N_TOK 4096
#define H_DIM 1024
#define T_EXP 8
#define E_DIM 512

// ---------------- GEMM tiling ----------------
#define BM 128
#define BN 128
#define BK 32
#define LDSTR 40   // ushort stride: 80B rows -> 16B aligned, bank-conflict-free

typedef __attribute__((ext_vector_type(16))) __bf16 v16bf;
typedef __attribute__((ext_vector_type(8)))  float  v8f;
typedef __attribute__((ext_vector_type(4)))  unsigned v4u;
typedef __attribute__((ext_vector_type(8)))  int      v8i;
typedef __attribute__((ext_vector_type(4)))  int      v4i;

union FragAB { v16bf v; uint2 q[4]; };
union FragC  { v8f   v; float f[8]; };

#if __has_builtin(__builtin_amdgcn_tensor_load_to_lds)
#define HAS_TDM 1
#else
#define HAS_TDM 0
#endif

__device__ __forceinline__ unsigned short f2bf(float f) {
    union { float f; unsigned u; } x; x.f = f;
    unsigned r = x.u + 0x7FFFu + ((x.u >> 16) & 1u);   // round-to-nearest-even
    return (unsigned short)(r >> 16);
}
__device__ __forceinline__ unsigned pack2(float lo, float hi) {
#if __has_builtin(__builtin_amdgcn_cvt_pk_bf16_f32)
    auto rv = __builtin_amdgcn_cvt_pk_bf16_f32(lo, hi);  // v_cvt_pk_bf16_f32
    unsigned u; __builtin_memcpy(&u, &rv, 4);
    return u;
#else
    return ((unsigned)f2bf(hi) << 16) | (unsigned)f2bf(lo);
#endif
}

#if HAS_TDM
// Issue one TDM 2D-tile load: 32(k) x 128(n) bf16, row stride = strideElems,
// destination LDS rows padded 64B data + 16B pad (pad_interval=16dw, amount=4dw).
__device__ __forceinline__ void tdm_issue_b(unsigned lds_addr,
                                            const unsigned short* gptr,
                                            unsigned strideElems)
{
    const unsigned long long ga = (unsigned long long)gptr;
    v4u g0;
    g0[0] = 1u;                                                 // count=1
    g0[1] = lds_addr;                                           // lds_addr
    g0[2] = (unsigned)ga;                                       // addr[31:0]
    g0[3] = (unsigned)((ga >> 32) & 0x1FFFFFFu) | (2u << 30);   // addr[56:32]|type=2
    v8i g1;
    g1[0] = (int)((1u << 16) | (1u << 20) | (3u << 22) | (3u << 25));
            // data_size=2B | pad_enable | pad_interval=16dw | pad_amount=4dw
    g1[1] = (int)0xFFFF0000u;                 // tensor_dim0 = 0x00FFFFFF (lo16)
    g1[2] = (int)0xFFFF00FFu;                 // dim0 hi16 | tensor_dim1 lo16
    g1[3] = (int)(0x000000FFu | (BK << 16));  // dim1 hi16 | tile_dim0=32
    g1[4] = (int)(BM);                        // tile_dim1=128 | tile_dim2=0
    g1[5] = (int)strideElems;                 // tensor_dim0_stride lo32
    g1[6] = 0;
    g1[7] = 0;
    const v4i z4 = {0, 0, 0, 0};
#if __clang_major__ >= 23
    const v8i z8 = {0, 0, 0, 0, 0, 0, 0, 0};
    __builtin_amdgcn_tensor_load_to_lds(g0, g1, z4, z4, z8, 0);
#else
    __builtin_amdgcn_tensor_load_to_lds(g0, g1, z4, z4, 0);
#endif
}
#endif

// ---------------- workspace layout (bytes) ----------------
#define WS_COUNTS 0                    // int[8]
#define WS_GATE   1024                 // float[N_TOK]
#define WS_BUCKET 17408                // int[T_EXP*N_TOK]
#define WS_ACT    148480               // bf16[N_TOK*E_DIM]           (4 MB)
#define WS_XBF    4342784              // bf16[N_TOK*H_DIM]           (8 MB)
#define WS_UPT    12731392             // bf16[T][E_DIM][H_DIM] n-major (8 MB)
#define WS_DNT    21120000             // bf16[T][H_DIM][E_DIM] n-major (8 MB)

// =====================================================================
// Kernel 1: router. One wave32 per token: 8 router logits + 1 gate dot.
// =====================================================================
__global__ __launch_bounds__(256) void router_kernel(
    const float* __restrict__ x, const float* __restrict__ router_w,
    const float* __restrict__ gate_w, float* __restrict__ gate_out,
    int* __restrict__ counts, int* __restrict__ bucket, int* __restrict__ ids_out)
{
    const int wave = threadIdx.x >> 5;
    const int lane = threadIdx.x & 31;
    const int n = blockIdx.x * 8 + wave;
    if (n >= N_TOK) return;

    const float* xr = x + (size_t)n * H_DIM;
    float acc[9];
#pragma unroll
    for (int t = 0; t < 9; ++t) acc[t] = 0.f;

    for (int i = 0; i < H_DIM / 128; ++i) {
        const int h = i * 128 + lane * 4;
        const float4 xv = *(const float4*)(xr + h);
#pragma unroll
        for (int t = 0; t < T_EXP; ++t) {
            const float4 wv = *(const float4*)(router_w + (size_t)t * H_DIM + h);
            acc[t] += xv.x * wv.x + xv.y * wv.y + xv.z * wv.z + xv.w * wv.w;
        }
        const float4 gv = *(const float4*)(gate_w + h);
        acc[8] += xv.x * gv.x + xv.y * gv.y + xv.z * gv.z + xv.w * gv.w;
    }
#pragma unroll
    for (int off = 16; off > 0; off >>= 1) {
#pragma unroll
        for (int t = 0; t < 9; ++t) acc[t] += __shfl_xor(acc[t], off, 32);
    }
    if (lane == 0) {
        int best = 0; float bv = acc[0];
#pragma unroll
        for (int t = 1; t < T_EXP; ++t) if (acc[t] > bv) { bv = acc[t]; best = t; }
        ids_out[n]  = best;
        gate_out[n] = 1.f / (1.f + __expf(-acc[8]));
        const int pos = atomicAdd(&counts[best], 1);
        bucket[best * N_TOK + pos] = n;
    }
}

// =====================================================================
// Kernel 2: elementwise f32 -> bf16 (x matrix), 8 elems/thread.
// =====================================================================
__global__ __launch_bounds__(256) void bf16_convert_kernel(
    const float* __restrict__ src, unsigned short* __restrict__ dst, int nElem)
{
    const int i = (blockIdx.x * 256 + threadIdx.x) * 8;
    if (i >= nElem) return;
    const float4 a = *(const float4*)(src + i);
    const float4 b = *(const float4*)(src + i + 4);
    uint4 o;
    o.x = pack2(a.x, a.y); o.y = pack2(a.z, a.w);
    o.z = pack2(b.x, b.y); o.w = pack2(b.z, b.w);
    *(uint4*)(dst + i) = o;
}

// =====================================================================
// Kernel 3: per-expert weight transpose + bf16:  dst[e][n][k] = src[e][k][n]
// =====================================================================
__global__ __launch_bounds__(256) void transpose_bf16_kernel(
    const float* __restrict__ src, unsigned short* __restrict__ dst,
    int K, int N)
{
    __shared__ unsigned short tile[32][33];
    const int e  = blockIdx.z;
    const int k0 = blockIdx.x * 32, n0 = blockIdx.y * 32;
    const int c  = threadIdx.x & 31, r8 = threadIdx.x >> 5;
    const float* s = src + (size_t)e * K * N;
#pragma unroll
    for (int ii = 0; ii < 4; ++ii) {
        const int k = k0 + r8 + ii * 8;
        tile[c][r8 + ii * 8] = f2bf(s[(size_t)k * N + n0 + c]);  // tile[n][k]
    }
    __syncthreads();
    unsigned short* d = dst + (size_t)e * N * K;
#pragma unroll
    for (int jj = 0; jj < 4; ++jj) {
        const int n = r8 + jj * 8;
        d[(size_t)(n0 + n) * K + k0 + c] = tile[n][c];
    }
}

// =====================================================================
// Grouped GEMM kernels. A: gathered bf16 rows (LDS staged per lane).
// B: TDM 2D tile -> LDS (double-buffered, overlapped with WMMA compute).
// =====================================================================

// ---- Kernel 4: up-GEMM + silu ----
__global__ __launch_bounds__(256) void up_kernel(
    const unsigned short* __restrict__ xbf, const unsigned short* __restrict__ upT,
    const int* __restrict__ counts, const int* __restrict__ bucket,
    unsigned short* __restrict__ act)
{
    __shared__ __attribute__((aligned(16))) unsigned short As[2][BM * LDSTR];
    __shared__ __attribute__((aligned(16))) unsigned short Bs[2][BN * LDSTR];

    const int e   = blockIdx.z;
    const int cnt = counts[e];
    const int m0  = blockIdx.x * BM;
    if (m0 >= cnt) return;
    const int n0  = blockIdx.y * BN;
    const int* buck = bucket + e * N_TOK;

    const int tid = threadIdx.x, lane = tid & 31, wave = tid >> 5;
    const int wm = wave & 3, wn = wave >> 2;

    FragC c[2][4];
#pragma unroll
    for (int mt = 0; mt < 2; ++mt)
#pragma unroll
        for (int nt = 0; nt < 4; ++nt)
#pragma unroll
            for (int r = 0; r < 8; ++r) c[mt][nt].f[r] = 0.f;

    const int ar = tid >> 1, ah = tid & 1;         // 2 threads / 32-elem row
    const int sA = m0 + ar;
    const int tokA = (sA < cnt) ? buck[sA] : -1;
    const unsigned short* arow = xbf + (size_t)(tokA < 0 ? 0 : tokA) * H_DIM + ah * 16;
    const unsigned short* bbase = upT + ((size_t)e * E_DIM + n0) * H_DIM;

    const int h = lane >> 4, ml = lane & 15;

    auto stageA = [&](int kb, int buf) {
        __builtin_prefetch(arow + kb + BK, 0, 3);
#pragma unroll
        for (int j = 0; j < 2; ++j) {
            uint4 v = make_uint4(0u, 0u, 0u, 0u);
            if (tokA >= 0) v = *(const uint4*)(arow + kb + j * 8);
            *(uint4*)&As[buf][ar * LDSTR + ah * 16 + j * 8] = v;
        }
    };
    auto compute = [&](int p) {
        FragAB a[2], b[4];
#pragma unroll
        for (int mt = 0; mt < 2; ++mt) {
            const unsigned short* q = &As[p][(wm * 32 + mt * 16 + ml) * LDSTR];
            a[mt].q[0] = *(const uint2*)(q + 16 * h);
            a[mt].q[1] = *(const uint2*)(q + 16 * h + 4);
            a[mt].q[2] = *(const uint2*)(q + 16 * h + 8);
            a[mt].q[3] = *(const uint2*)(q + 16 * h + 12);
        }
#pragma unroll
        for (int nt = 0; nt < 4; ++nt) {
            const unsigned short* q = &Bs[p][(wn * 64 + nt * 16 + ml) * LDSTR + 16 * h];
            b[nt].q[0] = *(const uint2*)(q);
            b[nt].q[1] = *(const uint2*)(q + 4);
            b[nt].q[2] = *(const uint2*)(q + 8);
            b[nt].q[3] = *(const uint2*)(q + 12);
        }
#pragma unroll
        for (int mt = 0; mt < 2; ++mt)
#pragma unroll
            for (int nt = 0; nt < 4; ++nt)
                c[mt][nt].v = __builtin_amdgcn_wmma_f32_16x16x32_bf16(
                    false, a[mt].v, false, b[nt].v, (short)0, c[mt][nt].v, false, false);
    };

    const int NIT = H_DIM / BK;
#if HAS_TDM
    const unsigned ldsB0 = (unsigned)(unsigned long long)&Bs[0][0];
    const unsigned ldsB1 = (unsigned)(unsigned long long)&Bs[1][0];
    stageA(0, 0);
    if (wave == 0) tdm_issue_b(ldsB0, bbase, H_DIM);
    for (int i = 0; i < NIT - 1; ++i) {
        const int p = i & 1, q = p ^ 1;
        stageA((i + 1) * BK, q);
        if (wave == 0) {
            tdm_issue_b(q ? ldsB1 : ldsB0, bbase + (i + 1) * BK, H_DIM);
            __builtin_amdgcn_s_wait_tensorcnt(1);   // tile i landed; i+1 in flight
        }
        __syncthreads();
        compute(p);
        __syncthreads();
    }
    if (wave == 0) __builtin_amdgcn_s_wait_tensorcnt(0);
    __syncthreads();
    compute((NIT - 1) & 1);
#else
    const unsigned short* brow = bbase + (size_t)ar * H_DIM + ah * 16;
    for (int i = 0; i < NIT; ++i) {
        stageA(i * BK, 0);
#pragma unroll
        for (int j = 0; j < 2; ++j)
            *(uint4*)&Bs[0][ar * LDSTR + ah * 16 + j * 8] =
                *(const uint4*)(brow + i * BK + j * 8);
        __syncthreads();
        compute(0);
        __syncthreads();
    }
#endif

#pragma unroll
    for (int mt = 0; mt < 2; ++mt) {
        const int srow = m0 + wm * 32 + mt * 16 + 8 * h;
#pragma unroll
        for (int nt = 0; nt < 4; ++nt) {
            const int col = n0 + wn * 64 + nt * 16 + ml;
#pragma unroll
            for (int r = 0; r < 8; ++r) {
                const int s = srow + r;
                if (s < cnt) {
                    const int tok = buck[s];
                    const float v = c[mt][nt].f[r];
                    act[(size_t)tok * E_DIM + col] = f2bf(v / (1.f + __expf(-v)));
                }
            }
        }
    }
}

// ---- Kernel 5: down-GEMM + gated residual ----
__global__ __launch_bounds__(256) void down_kernel(
    const float* __restrict__ x, const unsigned short* __restrict__ dnT,
    const int* __restrict__ counts, const int* __restrict__ bucket,
    const unsigned short* __restrict__ act, const float* __restrict__ gate,
    float* __restrict__ out)
{
    __shared__ __attribute__((aligned(16))) unsigned short As[2][BM * LDSTR];
    __shared__ __attribute__((aligned(16))) unsigned short Bs[2][BN * LDSTR];

    const int e   = blockIdx.z;
    const int cnt = counts[e];
    const int m0  = blockIdx.x * BM;
    if (m0 >= cnt) return;
    const int n0  = blockIdx.y * BN;
    const int* buck = bucket + e * N_TOK;

    const int tid = threadIdx.x, lane = tid & 31, wave = tid >> 5;
    const int wm = wave & 3, wn = wave >> 2;

    FragC c[2][4];
#pragma unroll
    for (int mt = 0; mt < 2; ++mt)
#pragma unroll
        for (int nt = 0; nt < 4; ++nt)
#pragma unroll
            for (int r = 0; r < 8; ++r) c[mt][nt].f[r] = 0.f;

    const int ar = tid >> 1, ah = tid & 1;
    const int sA = m0 + ar;
    const int tokA = (sA < cnt) ? buck[sA] : -1;
    const unsigned short* arow = act + (size_t)(tokA < 0 ? 0 : tokA) * E_DIM + ah * 16;
    const unsigned short* bbase = dnT + ((size_t)e * H_DIM + n0) * E_DIM;

    const int h = lane >> 4, ml = lane & 15;

    auto stageA = [&](int kb, int buf) {
        __builtin_prefetch(arow + kb + BK, 0, 3);
#pragma unroll
        for (int j = 0; j < 2; ++j) {
            uint4 v = make_uint4(0u, 0u, 0u, 0u);
            if (tokA >= 0) v = *(const uint4*)(arow + kb + j * 8);
            *(uint4*)&As[buf][ar * LDSTR + ah * 16 + j * 8] = v;
        }
    };
    auto compute = [&](int p) {
        FragAB a[2], b[4];
#pragma unroll
        for (int mt = 0; mt < 2; ++mt) {
            const unsigned short* q = &As[p][(wm * 32 + mt * 16 + ml) * LDSTR];
            a[mt].q[0] = *(const uint2*)(q + 16 * h);
            a[mt].q[1] = *(const uint2*)(q + 16 * h + 4);
            a[mt].q[2] = *(const uint2*)(q + 16 * h + 8);
            a[mt].q[3] = *(const uint2*)(q + 16 * h + 12);
        }
#pragma unroll
        for (int nt = 0; nt < 4; ++nt) {
            const unsigned short* q = &Bs[p][(wn * 64 + nt * 16 + ml) * LDSTR + 16 * h];
            b[nt].q[0] = *(const uint2*)(q);
            b[nt].q[1] = *(const uint2*)(q + 4);
            b[nt].q[2] = *(const uint2*)(q + 8);
            b[nt].q[3] = *(const uint2*)(q + 12);
        }
#pragma unroll
        for (int mt = 0; mt < 2; ++mt)
#pragma unroll
            for (int nt = 0; nt < 4; ++nt)
                c[mt][nt].v = __builtin_amdgcn_wmma_f32_16x16x32_bf16(
                    false, a[mt].v, false, b[nt].v, (short)0, c[mt][nt].v, false, false);
    };

    const int NIT = E_DIM / BK;
#if HAS_TDM
    const unsigned ldsB0 = (unsigned)(unsigned long long)&Bs[0][0];
    const unsigned ldsB1 = (unsigned)(unsigned long long)&Bs[1][0];
    stageA(0, 0);
    if (wave == 0) tdm_issue_b(ldsB0, bbase, E_DIM);
    for (int i = 0; i < NIT - 1; ++i) {
        const int p = i & 1, q = p ^ 1;
        stageA((i + 1) * BK, q);
        if (wave == 0) {
            tdm_issue_b(q ? ldsB1 : ldsB0, bbase + (i + 1) * BK, E_DIM);
            __builtin_amdgcn_s_wait_tensorcnt(1);
        }
        __syncthreads();
        compute(p);
        __syncthreads();
    }
    if (wave == 0) __builtin_amdgcn_s_wait_tensorcnt(0);
    __syncthreads();
    compute((NIT - 1) & 1);
#else
    const unsigned short* brow = bbase + (size_t)ar * E_DIM + ah * 16;
    for (int i = 0; i < NIT; ++i) {
        stageA(i * BK, 0);
#pragma unroll
        for (int j = 0; j < 2; ++j)
            *(uint4*)&Bs[0][ar * LDSTR + ah * 16 + j * 8] =
                *(const uint4*)(brow + i * BK + j * 8);
        __syncthreads();
        compute(0);
        __syncthreads();
    }
#endif

#pragma unroll
    for (int mt = 0; mt < 2; ++mt) {
        const int srow = m0 + wm * 32 + mt * 16 + 8 * h;
#pragma unroll
        for (int nt = 0; nt < 4; ++nt) {
            const int col = n0 + wn * 64 + nt * 16 + ml;
#pragma unroll
            for (int r = 0; r < 8; ++r) {
                const int s = srow + r;
                if (s < cnt) {
                    const int tok = buck[s];
                    const size_t idx = (size_t)tok * H_DIM + col;
                    out[idx] = x[idx] + gate[tok] * c[mt][nt].f[r];
                }
            }
        }
    }
}

// =====================================================================
extern "C" void kernel_launch(void* const* d_in, const int* in_sizes, int n_in,
                              void* d_out, int out_size, void* d_ws, size_t ws_size,
                              hipStream_t stream) {
    const float* x         = (const float*)d_in[0];
    const float* up_proj   = (const float*)d_in[1];
    const float* down_proj = (const float*)d_in[2];
    const float* router_w  = (const float*)d_in[3];
    const float* gate_w    = (const float*)d_in[4];

    float* out = (float*)d_out;
    int*   ids = (int*)d_out + (size_t)N_TOK * H_DIM;   // tuple output #2 (int32)

    char* ws = (char*)d_ws;
    int*            counts = (int*)(ws + WS_COUNTS);
    float*          gate   = (float*)(ws + WS_GATE);
    int*            bucket = (int*)(ws + WS_BUCKET);
    unsigned short* act    = (unsigned short*)(ws + WS_ACT);
    unsigned short* xbf    = (unsigned short*)(ws + WS_XBF);
    unsigned short* upT    = (unsigned short*)(ws + WS_UPT);
    unsigned short* dnT    = (unsigned short*)(ws + WS_DNT);

    (void)hipMemsetAsync(counts, 0, T_EXP * sizeof(int), stream);

    router_kernel<<<N_TOK / 8, 256, 0, stream>>>(x, router_w, gate_w,
                                                 gate, counts, bucket, ids);

    bf16_convert_kernel<<<(N_TOK * H_DIM) / (256 * 8), 256, 0, stream>>>(
        x, xbf, N_TOK * H_DIM);

    transpose_bf16_kernel<<<dim3(H_DIM / 32, E_DIM / 32, T_EXP), 256, 0, stream>>>(
        up_proj, upT, H_DIM, E_DIM);
    transpose_bf16_kernel<<<dim3(E_DIM / 32, H_DIM / 32, T_EXP), 256, 0, stream>>>(
        down_proj, dnT, E_DIM, H_DIM);

    dim3 gUp(N_TOK / BM, E_DIM / BN, T_EXP);     // (32, 4, 8) worst-case grid
    up_kernel<<<gUp, 256, 0, stream>>>(xbf, upT, counts, bucket, act);

    dim3 gDn(N_TOK / BM, H_DIM / BN, T_EXP);     // (32, 8, 8)
    down_kernel<<<gDn, 256, 0, stream>>>(x, dnT, counts, bucket,
                                         act, gate, out);
}
